// CrfRnn_55825984914030
// MI455X (gfx1250) — matrix operand order repaired and checked by reading, not compiled, not run on hardware
//
#include <hip/hip_runtime.h>

// CRF-RNN mean-field inference, MI455X (gfx1250, wave32, WMMA).
// - Bilateral branch of the reference is dead (message uses spatial_out twice),
//   so rgb / Kb are never computed.
// - Spatial Gaussian is separable: p @ Ks == GEMM with Gx (128x128) along x,
//   then Gy (96x96) along y; norm_s = sx[x]*sy[y]. ~58 M MACs/iter instead of
//   6.3 GFLOP/iter dense + a 604 MB kernel matrix -> latency-bound.
// - Blur GEMMs use v_wmma_f32_16x16x32_f16. All fragment loads are 16-byte
//   global_load_b128: A rows are contiguous per the ISA A-layout; B columns are
//   contiguous because Gx is symmetric and the intermediate is stored
//   transposed (T1t[c][w][h]), which the x-blur C-fragment emits as one
//   contiguous v8h store per lane.

typedef __attribute__((ext_vector_type(16))) _Float16 v16h;
typedef __attribute__((ext_vector_type(8)))  _Float16 v8h;
typedef __attribute__((ext_vector_type(8)))  float    v8f;

constexpr int HH = 96;
constexpr int WW = 128;
constexpr int CC = 21;
constexpr int NN = HH * WW;        // 12288 pixels
constexpr int RR = CC * HH;        // 2016 rows for the x-blur GEMM (126 tiles)
constexpr float INV_THETA_G = 1.0f / 3.0f;

// ---------------- WMMA fragment loaders (wave32, 16x16x32 f16) --------------
// A (16x32, MxK), row-major source: lane holds row M = lane%16; its 16 halfs
// are K = {kb..kb+7, kb+16..kb+23}, kb = 8*(lane/16)  -> two b128 loads.
__device__ inline v16h ld_frag_A_rows(const _Float16* __restrict__ A, int lda,
                                      int row0, int col0) {
  const int lane = threadIdx.x & 31;
  const int m    = lane & 15;
  const int kb   = (lane >> 4) << 3;               // 0 or 8
  const _Float16* p = A + (size_t)(row0 + m) * lda + col0 + kb;
  v8h lo = *(const v8h*)(p);                       // K = kb .. kb+7
  v8h hi = *(const v8h*)(p + 16);                  // K = kb+16 .. kb+23
  return __builtin_shufflevector(lo, hi, 0, 1, 2, 3, 4, 5, 6, 7,
                                         8, 9, 10, 11, 12, 13, 14, 15);
}

// B (32x16, KxN) from a COLUMN-CONTIGUOUS source: column n lives at
// Bcol + (col0+n)*ldb, K contiguous. Lane holds column N = lane%16,
// K = {kb..kb+7, kb+16..kb+23}, kb = 8*(lane/16)  -> two b128 loads.
__device__ inline v16h ld_frag_B_cols(const _Float16* __restrict__ Bcol, int ldb,
                                      int row0, int col0) {
  const int lane = threadIdx.x & 31;
  const int n    = lane & 15;
  const int kb   = (lane >> 4) << 3;               // 0 or 8
  const _Float16* p = Bcol + (size_t)(col0 + n) * ldb + row0 + kb;
  v8h lo = *(const v8h*)(p);                       // K = kb .. kb+7
  v8h hi = *(const v8h*)(p + 16);                  // K = kb+16 .. kb+23
  return __builtin_shufflevector(lo, hi, 0, 1, 2, 3, 4, 5, 6, 7,
                                         8, 9, 10, 11, 12, 13, 14, 15);
}

// ---------------- setup kernels ---------------------------------------------
// Transpose unaries NHWC -> [C][N]; initialize q = u.
__global__ void prep_kernel(const float* __restrict__ un,
                            float* __restrict__ U, float* __restrict__ Q) {
  int idx = blockIdx.x * blockDim.x + threadIdx.x;
  if (idx >= CC * NN) return;
  int c = idx / NN, n = idx - c * NN;
  float v = un[(size_t)n * CC + c];
  U[idx] = v;
  Q[idx] = v;
}

// Gaussian tables Gx[128x128], Gy[96x96] in f16 + row sums computed from the
// f16-rounded entries so normalization matches the WMMA blur exactly.
__global__ void tables_kernel(_Float16* __restrict__ Gx, _Float16* __restrict__ Gy,
                              float* __restrict__ sx, float* __restrict__ sy) {
  int i = threadIdx.x;
  if (i < WW) {
    float s = 0.f;
    for (int j = 0; j < WW; ++j) {
      float d = (float)(i - j) * INV_THETA_G;
      _Float16 g = (_Float16)__expf(-0.5f * d * d);
      Gx[i * WW + j] = g;
      s += (float)g;
    }
    sx[i] = s;
  }
  if (i < HH) {
    float s = 0.f;
    for (int j = 0; j < HH; ++j) {
      float d = (float)(i - j) * INV_THETA_G;
      _Float16 g = (_Float16)__expf(-0.5f * d * d);
      Gy[i * HH + j] = g;
      s += (float)g;
    }
    sy[i] = s;
  }
}

// A2 = comp @ (sw + bw)  (21x21), folding the replicated-spatial-term bug.
__global__ void mix_kernel(const float* __restrict__ sw, const float* __restrict__ bw,
                           const float* __restrict__ comp, float* __restrict__ A2) {
  int idx = threadIdx.x;
  if (idx >= CC * CC) return;
  int i = idx / CC, j = idx - i * CC;
  float acc = 0.f;
  for (int k = 0; k < CC; ++k)
    acc += comp[i * CC + k] * (sw[k * CC + j] + bw[k * CC + j]);
  A2[idx] = acc;
}

// ---------------- per-iteration kernels -------------------------------------
// Softmax over classes at each pixel; emit p as f16 for the WMMA blur.
__global__ void softmax_kernel(const float* __restrict__ Q, _Float16* __restrict__ P) {
  int n = blockIdx.x * blockDim.x + threadIdx.x;
  if (n >= NN) return;
  float v[CC];
  float mx = -3.4e38f;
#pragma unroll
  for (int c = 0; c < CC; ++c) { v[c] = Q[c * NN + n]; mx = fmaxf(mx, v[c]); }
  float s = 0.f;
#pragma unroll
  for (int c = 0; c < CC; ++c) { v[c] = __expf(v[c] - mx); s += v[c]; }
  float inv = 1.0f / s;
#pragma unroll
  for (int c = 0; c < CC; ++c) P[c * NN + n] = (_Float16)(v[c] * inv);
}

// X-blur: T1[r, w] = sum_{w'} P[r, w'] * Gx[w', w]; r = c*H + h.
// GEMM [2016x128]x[128x128]; 8 waves/block, 1 tile/wave, K = 4 chained WMMAs.
// Gx is symmetric, so B columns are its (contiguous) rows. The result is
// stored TRANSPOSED as T1t[c][w][h]: each lane holds 8 consecutive h at fixed
// w -> a single contiguous v8h (16 B) store.
__global__ void __launch_bounds__(256)
xblur_wmma(const _Float16* __restrict__ P, const _Float16* __restrict__ Gx,
           _Float16* __restrict__ T1t) {
  const int tile = blockIdx.x * 8 + (threadIdx.x >> 5);   // 0..1007
  const int nt = tile & 7;                                // w tile
  const int mt = tile >> 3;                               // row tile (0..125)
  v8f acc = {};
#pragma unroll
  for (int kk = 0; kk < WW / 32; ++kk) {
    v16h a = ld_frag_A_rows(P,  WW, mt * 16, kk * 32);
    v16h b = ld_frag_B_cols(Gx, WW, kk * 32, nt * 16);    // symmetric: cols==rows
    acc = __builtin_amdgcn_wmma_f32_16x16x32_f16(false, a, false, b,
                                                 (short)0, acc, false, false);
  }
  const int lane = threadIdx.x & 31;
  const int w  = nt * 16 + (lane & 15);
  const int r0 = mt * 16 + ((lane >> 4) << 3);            // 8 consecutive rows
  const int c  = r0 / HH;                                 // tile never straddles c
  const int h0 = r0 - c * HH;
  v8h o;
#pragma unroll
  for (int v = 0; v < 8; ++v) o[v] = (_Float16)acc[v];
  *(v8h*)(T1t + ((size_t)c * WW + w) * HH + h0) = o;
}

// Y-blur + normalization: out2[c,h,w] = (sum_{h'} Gy[h,h'] T1[c,h',w]) / (sx[w]*sy[h]).
// Per class: GEMM [96x96]x[96x128]. B columns (fixed w, varying h') are
// contiguous in T1t. 8 waves/block, K = 3 chained WMMAs.
__global__ void __launch_bounds__(256)
yblur_wmma(const _Float16* __restrict__ Gy, const _Float16* __restrict__ T1t,
           const float* __restrict__ sx, const float* __restrict__ sy,
           float* __restrict__ out2) {
  const int tile = blockIdx.x * 8 + (threadIdx.x >> 5);   // 0..1007
  const int nt   = tile & 7;                              // w tile
  const int rest = tile >> 3;
  const int mt   = rest % 6;                              // h tile
  const int c    = rest / 6;                              // class
  const _Float16* Bc = T1t + (size_t)c * NN;              // [w][h] column-contig
  v8f acc = {};
#pragma unroll
  for (int kk = 0; kk < HH / 32; ++kk) {
    v16h a = ld_frag_A_rows(Gy, HH, mt * 16, kk * 32);
    v16h b = ld_frag_B_cols(Bc, HH, kk * 32, nt * 16);
    acc = __builtin_amdgcn_wmma_f32_16x16x32_f16(false, a, false, b,
                                                 (short)0, acc, false, false);
  }
  const int lane = threadIdx.x & 31;
  const int w  = nt * 16 + (lane & 15);
  const int mb = (lane >> 4) << 3;
  const float isx = 1.0f / sx[w];
#pragma unroll
  for (int v = 0; v < 8; ++v) {
    const int h = mt * 16 + mb + v;
    out2[(size_t)c * NN + h * WW + w] = acc[v] * isx / sy[h];
  }
}

// q = u - A2 @ spatial_out  (441 VALU MACs per pixel, fully coalesced).
__global__ void update_kernel(const float* __restrict__ U,
                              const float* __restrict__ out2,
                              const float* __restrict__ A2,
                              float* __restrict__ Q) {
  __shared__ float sA[CC * CC];
  for (int i = threadIdx.x; i < CC * CC; i += blockDim.x) sA[i] = A2[i];
  __syncthreads();
  int n = blockIdx.x * blockDim.x + threadIdx.x;
  if (n >= NN) return;
  float s[CC];
#pragma unroll
  for (int k = 0; k < CC; ++k) s[k] = out2[k * NN + n];
#pragma unroll
  for (int c = 0; c < CC; ++c) {
    float acc = 0.f;
#pragma unroll
    for (int k = 0; k < CC; ++k) acc = fmaf(sA[c * CC + k], s[k], acc);
    Q[c * NN + n] = U[c * NN + n] - acc;
  }
}

// (1,C,H,W) -> (1,W,H,C):  out[(w*H + h)*C + c] = Q[c][h*W + w]
__global__ void output_kernel(const float* __restrict__ Q, float* __restrict__ out) {
  int idx = blockIdx.x * blockDim.x + threadIdx.x;
  if (idx >= CC * NN) return;
  int c = idx % CC;
  int rem = idx / CC;
  int h = rem % HH;
  int w = rem / HH;
  out[idx] = Q[c * NN + h * WW + w];
}

// ---------------- host-side launch ------------------------------------------
extern "C" void kernel_launch(void* const* d_in, const int* in_sizes, int n_in,
                              void* d_out, int out_size, void* d_ws, size_t ws_size,
                              hipStream_t stream) {
  const float* un   = (const float*)d_in[0];   // unaries (1,H,W,C)
  // d_in[1] (rgb) is dead: the source uses spatial_out for BOTH message terms.
  const float* sw   = (const float*)d_in[2];   // spatial_ker_weights (C,C)
  const float* bw   = (const float*)d_in[3];   // bilateral_ker_weights (C,C)
  const float* comp = (const float*)d_in[4];   // compatibility_matrix (C,C)
  float* out = (float*)d_out;

  char* ws = (char*)d_ws;
  size_t off = 0;
  auto carve = [&](size_t bytes) -> void* {
    void* p = ws + off;
    off = (off + bytes + 255) & ~(size_t)255;
    return p;
  };
  float*    U    = (float*)   carve(sizeof(float)    * CC * NN);
  float*    Q    = (float*)   carve(sizeof(float)    * CC * NN);
  float*    out2 = (float*)   carve(sizeof(float)    * CC * NN);
  _Float16* P    = (_Float16*)carve(sizeof(_Float16) * CC * NN);
  _Float16* T1t  = (_Float16*)carve(sizeof(_Float16) * CC * NN);  // [C][W][H]
  _Float16* Gx   = (_Float16*)carve(sizeof(_Float16) * WW * WW);
  _Float16* Gy   = (_Float16*)carve(sizeof(_Float16) * HH * HH);
  float*    sx   = (float*)   carve(sizeof(float)    * WW);
  float*    sy   = (float*)   carve(sizeof(float)    * HH);
  float*    A2   = (float*)   carve(sizeof(float)    * CC * CC);
  (void)ws_size; (void)in_sizes; (void)n_in; (void)out_size;   // ~4.2 MB used

  prep_kernel  <<<(CC * NN + 255) / 256, 256, 0, stream>>>(un, U, Q);
  tables_kernel<<<1, 128, 0, stream>>>(Gx, Gy, sx, sy);
  mix_kernel   <<<1, 512, 0, stream>>>(sw, bw, comp, A2);

  constexpr int XT = (RR / 16) * (WW / 16);        // 1008 x-blur tiles
  constexpr int YT = CC * (HH / 16) * (WW / 16);   // 1008 y-blur tiles
  for (int it = 0; it < 5; ++it) {
    softmax_kernel<<<(NN + 255) / 256, 256, 0, stream>>>(Q, P);
    xblur_wmma<<<XT / 8, 256, 0, stream>>>(P, Gx, T1t);
    yblur_wmma<<<YT / 8, 256, 0, stream>>>(Gy, T1t, sx, sy, out2);
    update_kernel<<<(NN + 255) / 256, 256, 0, stream>>>(U, out2, A2, Q);
  }

  output_kernel<<<(CC * NN + 255) / 256, 256, 0, stream>>>(Q, out);
}
